// MPNN_67095979098696
// MI455X (gfx1250) — compile-verified, matrix operand order checked by reading, not run on hardware
//
#include <hip/hip_runtime.h>
#include <hip/hip_bf16.h>

// ---------------- WMMA fragment helpers (CDNA5, wave32) ----------------
typedef __attribute__((ext_vector_type(16))) _Float16 v16h;
typedef __attribute__((ext_vector_type(8)))  float    v8f;

union FragH { _Float16 e[16]; v16h v; };
union FragF { float    e[8];  v8f  v; };

// 16-bit A/B fragment K-index for lane/half-slot p (ISA 7.12.2, 16-bit A 16x32):
// VGPR r (=p/2): lanes 0-15 hold K = {0,1},{2,3},{4,5},{6,7} then +16 for r>=4;
// lanes 16-31 hold K offset by +8 within each 16-K half.
__device__ __forceinline__ int wmma_k(int lane, int p) {
  int r  = p >> 1;
  int lo = p & 1;
  int base = (lane & 16) ? 8 : 0;
  int k = (r < 4) ? (base + 2 * r) : (16 + base + 2 * (r - 4));
  return k + lo;
}

__device__ __forceinline__ v8f wmma16x16x32(const FragH& a, const FragH& b, v8f c) {
  return __builtin_amdgcn_wmma_f32_16x16x32_f16(false, a.v, false, b.v,
                                                (short)0, c, false, false);
}

// ---------------- constants ----------------
#define HDIM   64
#define FNODE  128
#define FEDGE  16
#define OUTD   32
#define STEPS  3
#define EW_STRIDE 4104   // 4096 + 8 halves pad: D-frag column stores hit distinct banks
#define EPB    32        // edges per message block (2 M-tiles share each B fetch)

// ------- weight preconversion: f32 -> f16 *fragment-packed* B layouts -------
// Packed layout: frag (tile t, kstep s) -> 32 lanes x 16 halves contiguous, so a
// lane's whole fragment is one aligned 32-byte load (global_load_b128 x2).
__global__ void prep_weights(const float* __restrict__ W_e2,
                             const float* __restrict__ W_enc,
                             const float* __restrict__ W_ih,
                             const float* __restrict__ W_hh,
                             _Float16* __restrict__ W2p,
                             _Float16* __restrict__ WencP,
                             _Float16* __restrict__ WihP,
                             _Float16* __restrict__ WhhP) {
  int i = blockIdx.x * blockDim.x + threadIdx.x;
  int p = i & 15, lane = (i >> 4) & 31, nl = lane & 15;
  if (i < 256 * 2 * 32 * 16) {                 // W_e2 [64,4096]: 256 n-tiles x 2 ksteps
    int s = (i >> 9) & 1, t = i >> 10;
    int k = s * 32 + wmma_k(lane, p);
    int n = t * 16 + nl;
    W2p[i] = (_Float16)W_e2[k * 4096 + n];
  }
  if (i < 4 * 4 * 32 * 16) {                   // W_enc [128,64]: 4 n-tiles x 4 ksteps
    int ks = (i >> 9) & 3, t = i >> 11;
    int k = ks * 32 + wmma_k(lane, p);
    int n = t * 16 + nl;
    WencP[i] = (_Float16)W_enc[k * 64 + n];
  }
  if (i < 12 * 2 * 32 * 16) {                  // W^T [64,192]: 12 n-tiles x 2 ksteps
    int s = (i >> 9) & 1, t = i >> 10;
    int kk = s * 32 + wmma_k(lane, p);
    int n = t * 16 + nl;
    WihP[i] = (_Float16)W_ih[n * 64 + kk];     // transpose folded in (torch [3H,H])
    WhhP[i] = (_Float16)W_hh[n * 64 + kk];
  }
}

// ---- z = relu(edge_attr @ W_e1 + b_e1), emitted directly in A-fragment order ----
// zp[((g*2+s)*32 + lane)*16 + p] = z[g*16 + (lane&15)][s*32 + wmma_k(lane,p)]
__global__ void edge_z(const float* __restrict__ ea, const float* __restrict__ W_e1,
                       const float* __restrict__ b_e1, _Float16* __restrict__ zp, int E_) {
  int idx = blockIdx.x * blockDim.x + threadIdx.x;
  if (idx >= E_ * HDIM) return;
  int p = idx & 15, lane = (idx >> 4) & 31, s = (idx >> 9) & 1;
  long g = idx >> 10;
  long e = g * 16 + (lane & 15);
  int i = s * 32 + wmma_k(lane, p);
  float acc = b_e1[i];
#pragma unroll
  for (int k = 0; k < FEDGE; ++k) acc += ea[e * FEDGE + k] * W_e1[k * HDIM + i];
  zp[idx] = (_Float16)fmaxf(acc, 0.f);
}

// ---------------- h0 = x @ W_enc + b_enc  (WMMA, 16 nodes per block) ----------------
__global__ __launch_bounds__(128) void encode(const float* __restrict__ x,
                                              const _Float16* __restrict__ WencP,
                                              const float* __restrict__ b_enc,
                                              float* __restrict__ h,
                                              _Float16* __restrict__ hh) {
  __shared__ _Float16 Xg[16 * FNODE];
  int tid = threadIdx.x;
  long node0 = (long)blockIdx.x * 16;
  for (int q = tid; q < 16 * FNODE; q += 128)
    Xg[q] = (_Float16)x[node0 * FNODE + q];
  __syncthreads();

  int lane = tid & 31, wave = tid >> 5;
  int nl = lane & 15;
  int mhalf = (lane & 16) ? 8 : 0;
  int n = wave * 16 + nl;                      // output column 0..63
  const v16h* Bv = (const v16h*)WencP;

  v8f acc = {};
#pragma unroll
  for (int ks = 0; ks < 4; ++ks) {
    FragH a, b;
#pragma unroll
    for (int p = 0; p < 16; ++p)
      a.e[p] = Xg[nl * FNODE + ks * 32 + wmma_k(lane, p)];
    b.v = Bv[(wave * 4 + ks) * 32 + lane];     // one 32B vector load
    acc = wmma16x16x32(a, b, acc);
  }
  float bias = b_enc[n];
  FragF d; d.v = acc;
#pragma unroll
  for (int r = 0; r < 8; ++r) {
    long m = node0 + mhalf + r;
    float v = d.e[r] + bias;
    h[m * HDIM + n] = v;
    hh[m * HDIM + n] = (_Float16)v;
  }
}

// ---------------- zero helper ----------------
__global__ void zero_f32(float* __restrict__ p, int n) {
  int i = blockIdx.x * blockDim.x + threadIdx.x;
  if (i < n) p[i] = 0.f;
}

// ---- message: per 32-edge block, EW[32,4096]=Z@W_e2+b in LDS (WMMA), then matvec+scatter ----
// Two 16-edge M-tiles share every B-fragment fetch: 4 WMMAs per B pair -> 2x L2 intensity.
__global__ __launch_bounds__(256) void message(const _Float16* __restrict__ zp,
                                               const _Float16* __restrict__ W2p,
                                               const float* __restrict__ b_e2,
                                               const _Float16* __restrict__ hh,
                                               const int* __restrict__ src,
                                               const int* __restrict__ dst,
                                               float* __restrict__ m_agg) {
  extern __shared__ char smem[];
  _Float16* EW = (_Float16*)smem;                               // EPB x EW_STRIDE f16
  float*    Hs = (float*)(smem + EPB * EW_STRIDE * 2);          // EPB x 64 f32
  int*      Dg = (int*)(smem + EPB * EW_STRIDE * 2 + EPB * HDIM * 4);

  int tid = threadIdx.x;
  long e0 = (long)blockIdx.x * EPB;

  // stage gathered h_src (f16 -> f32) and dst ids
  for (int q = tid; q < EPB * HDIM; q += 256) {
    int e = q >> 6, i = q & 63;
    Hs[q] = (float)hh[(long)src[e0 + e] * HDIM + i];
  }
  if (tid < EPB) Dg[tid] = dst[e0 + tid];

  int lane = tid & 31, wave = tid >> 5;
  int nl = lane & 15;
  int mhalf = (lane & 16) ? 8 : 0;

  // A fragments for the two 16-edge groups, direct 32B vector loads
  const v16h* Zv = (const v16h*)zp + (long)blockIdx.x * 128 + lane;
  FragH a0g0, a1g0, a0g1, a1g1;
  a0g0.v = Zv[0];          // group 2b+0, kstep 0
  a1g0.v = Zv[32];         // group 2b+0, kstep 1
  a0g1.v = Zv[64];         // group 2b+1, kstep 0
  a1g1.v = Zv[96];         // group 2b+1, kstep 1

  const v16h* Bv = (const v16h*)W2p;
  for (int t = wave; t < 256; t += 8) {        // 32 N-tiles per wave
    if (t + 8 < 256)                            // prefetch next tile's B frags
      __builtin_prefetch(Bv + (t + 8) * 64 + lane, 0, 1);
    FragH b0, b1;
    b0.v = Bv[t * 64 + lane];
    b1.v = Bv[t * 64 + 32 + lane];
    v8f acc0 = {}, acc1 = {};
    acc0 = wmma16x16x32(a0g0, b0, acc0);
    acc1 = wmma16x16x32(a0g1, b0, acc1);
    acc0 = wmma16x16x32(a1g0, b1, acc0);
    acc1 = wmma16x16x32(a1g1, b1, acc1);
    int n = t * 16 + nl;
    float bias = b_e2[n];
    FragF d0; d0.v = acc0;
    FragF d1; d1.v = acc1;
#pragma unroll
    for (int r = 0; r < 8; ++r) {
      EW[(mhalf + r) * EW_STRIDE + n]        = (_Float16)(d0.e[r] + bias);
      EW[(16 + mhalf + r) * EW_STRIDE + n]   = (_Float16)(d1.e[r] + bias);
    }
  }
  __syncthreads();   // single barrier: covers Hs/Dg staging and EW production

  // msg[e,i] = sum_j EW[e][i*64+j] * h_src[e][j]; scatter-add to m_agg[dst]
  // thread mapping e = q&31 spreads rows across LDS banks (row stride 4104)
  for (int q = tid; q < EPB * HDIM; q += 256) {
    int e = q & 31, i = q >> 5;
    const _Float16* row = EW + e * EW_STRIDE + i * 64;
    const float* hv = Hs + e * 64;
    float s = 0.f;
#pragma unroll
    for (int j = 0; j < 64; ++j) s += (float)row[j] * hv[j];
    atomicAdd(m_agg + (long)Dg[e] * HDIM + i, s);
  }
}

// ---------------- GRU cell: gi=m@W_ih^T+b_ih, gh=h@W_hh^T+b_hh (WMMA) + gates ----------------
__global__ __launch_bounds__(128) void gru(const float* __restrict__ m_agg,
                                           float* __restrict__ h,
                                           _Float16* __restrict__ hh,
                                           const _Float16* __restrict__ WihP,
                                           const _Float16* __restrict__ WhhP,
                                           const float* __restrict__ b_ih,
                                           const float* __restrict__ b_hh) {
  __shared__ _Float16 Mg[16 * HDIM];
  __shared__ _Float16 Hg[16 * HDIM];
  __shared__ float    G[16 * 384];             // gi: cols 0..191, gh: 192..383
  int tid = threadIdx.x;
  long n0 = (long)blockIdx.x * 16;

  for (int q = tid; q < 1024; q += 128) {
    Mg[q] = (_Float16)m_agg[n0 * HDIM + q];
    Hg[q] = hh[n0 * HDIM + q];
  }
  __syncthreads();

  int lane = tid & 31, wave = tid >> 5;
  int nl = lane & 15;
  int mhalf = (lane & 16) ? 8 : 0;

  FragH am0, am1, ah0, ah1;
#pragma unroll
  for (int p = 0; p < 16; ++p) {
    int k = wmma_k(lane, p);
    am0.e[p] = Mg[nl * HDIM + k];      am1.e[p] = Mg[nl * HDIM + 32 + k];
    ah0.e[p] = Hg[nl * HDIM + k];      ah1.e[p] = Hg[nl * HDIM + 32 + k];
  }

  for (int job = wave; job < 24; job += 4) {   // 12 gi tiles + 12 gh tiles
    int ish = (job >= 12) ? 1 : 0;
    int tcol = ish ? (job - 12) : job;
    int n = tcol * 16 + nl;                    // 0..191
    const v16h* Bv = (const v16h*)(ish ? WhhP : WihP);
    FragH b0, b1;
    b0.v = Bv[tcol * 64 + lane];
    b1.v = Bv[tcol * 64 + 32 + lane];
    v8f acc = {};
    acc = wmma16x16x32(ish ? ah0 : am0, b0, acc);
    acc = wmma16x16x32(ish ? ah1 : am1, b1, acc);
    float bias = ish ? b_hh[n] : b_ih[n];
    FragF d; d.v = acc;
    int base = ish ? 192 : 0;
#pragma unroll
    for (int r = 0; r < 8; ++r)
      G[(mhalf + r) * 384 + base + n] = d.e[r] + bias;
  }
  __syncthreads();

  for (int q = tid; q < 1024; q += 128) {
    int m = q >> 6, i = q & 63;
    const float* gi = G + m * 384;
    const float* gh = G + m * 384 + 192;
    float r  = 1.f / (1.f + __expf(-(gi[i] + gh[i])));
    float zz = 1.f / (1.f + __expf(-(gi[64 + i] + gh[64 + i])));
    float nn = tanhf(gi[128 + i] + r * gh[128 + i]);
    long idx = (n0 + m) * HDIM + i;
    float hold = h[idx];
    float hnew = (1.f - zz) * nn + zz * hold;
    h[idx]  = hnew;
    hh[idx] = (_Float16)hnew;
  }
}

// ---------------- graph pooling: g[batch[v]] += h[v] ----------------
__global__ void pool(const float* __restrict__ h, const int* __restrict__ batch,
                     float* __restrict__ g, int N_) {
  int idx = blockIdx.x * blockDim.x + threadIdx.x;
  if (idx >= N_ * HDIM) return;
  int node = idx >> 6, i = idx & 63;
  atomicAdd(g + (long)batch[node] * HDIM + i, h[idx]);
}

// ---------------- readout MLP on [G=64, 64] (tiny; single block) ----------------
__global__ __launch_bounds__(256) void readout(const float* __restrict__ g,
                                               const float* __restrict__ W_r1,
                                               const float* __restrict__ b_r1,
                                               const float* __restrict__ W_r2,
                                               const float* __restrict__ b_r2,
                                               float* __restrict__ out, int G_) {
  __shared__ float Gs[64 * 64];
  __shared__ float T1[64 * 64];
  int tid = threadIdx.x;
  for (int q = tid; q < G_ * 64; q += 256) Gs[q] = g[q];
  __syncthreads();
  for (int q = tid; q < G_ * 64; q += 256) {
    int gr = q >> 6, i = q & 63;
    float acc = b_r1[i];
    for (int k = 0; k < 64; ++k) acc += Gs[gr * 64 + k] * W_r1[k * 64 + i];
    T1[q] = fmaxf(acc, 0.f);
  }
  __syncthreads();
  for (int q = tid; q < G_ * OUTD; q += 256) {
    int gr = q >> 5, o = q & 31;
    float acc = b_r2[o];
    for (int k = 0; k < 64; ++k) acc += T1[gr * 64 + k] * W_r2[k * OUTD + o];
    out[q] = acc;
  }
}

// ---------------- host driver ----------------
extern "C" void kernel_launch(void* const* d_in, const int* in_sizes, int n_in,
                              void* d_out, int out_size, void* d_ws, size_t ws_size,
                              hipStream_t stream) {
  const float* x         = (const float*)d_in[0];
  const int*   edge_idx  = (const int*)d_in[1];
  const float* edge_attr = (const float*)d_in[2];
  const int*   batch     = (const int*)d_in[3];
  const float* W_enc     = (const float*)d_in[4];
  const float* b_enc     = (const float*)d_in[5];
  const float* W_e1      = (const float*)d_in[6];
  const float* b_e1      = (const float*)d_in[7];
  const float* W_e2      = (const float*)d_in[8];
  const float* b_e2      = (const float*)d_in[9];
  const float* W_ih      = (const float*)d_in[10];
  const float* W_hh      = (const float*)d_in[11];
  const float* b_ih      = (const float*)d_in[12];
  const float* b_hh      = (const float*)d_in[13];
  const float* W_r1      = (const float*)d_in[14];
  const float* b_r1      = (const float*)d_in[15];
  const float* W_r2      = (const float*)d_in[16];
  const float* b_r2      = (const float*)d_in[17];
  float* out = (float*)d_out;

  const int N = in_sizes[0] / FNODE;      // 50000
  const int E = in_sizes[2] / FEDGE;      // 100000
  const int G = out_size / OUTD;          // 64
  const int* src = edge_idx;              // edge_index[0]
  const int* dst = edge_idx + E;          // edge_index[1]

  // workspace carve-out (256B aligned)
  char* ws = (char*)d_ws;
  size_t off = 0;
  auto carve = [&](size_t bytes) {
    void* p = ws + off;
    off = (off + bytes + 255) & ~(size_t)255;
    return p;
  };
  float*    h     = (float*)   carve((size_t)N * HDIM * 4);
  _Float16* hh    = (_Float16*)carve((size_t)N * HDIM * 2);
  float*    m_agg = (float*)   carve((size_t)N * HDIM * 4);
  _Float16* zp    = (_Float16*)carve((size_t)E * HDIM * 2);
  _Float16* W2p   = (_Float16*)carve((size_t)64 * 4096 * 2);
  _Float16* WencP = (_Float16*)carve((size_t)128 * 64 * 2);
  _Float16* WihP  = (_Float16*)carve((size_t)64 * 192 * 2);
  _Float16* WhhP  = (_Float16*)carve((size_t)64 * 192 * 2);
  float*    gbuf  = (float*)   carve((size_t)G * HDIM * 4);
  (void)ws_size;

  // 1) weight conversions into fragment-packed layouts
  prep_weights<<<(64 * 4096 + 255) / 256, 256, 0, stream>>>(W_e2, W_enc, W_ih, W_hh,
                                                            W2p, WencP, WihP, WhhP);
  // 2) edge hidden z (loop-invariant), fragment-packed
  edge_z<<<(E * HDIM + 255) / 256, 256, 0, stream>>>(edge_attr, W_e1, b_e1, zp, E);
  // 3) node encoder
  encode<<<N / 16, 128, 0, stream>>>(x, WencP, b_enc, h, hh);

  // message LDS: EW(32*4104*2 = 262656) + Hs(8KB) + dst  ~= 271KB (<= 320KB/WG)
  const size_t MSG_LDS = EPB * EW_STRIDE * 2 + EPB * HDIM * 4 + 256;

  for (int s = 0; s < STEPS; ++s) {
    zero_f32<<<(N * HDIM + 255) / 256, 256, 0, stream>>>(m_agg, N * HDIM);
    message<<<E / EPB, 256, MSG_LDS, stream>>>(zp, W2p, b_e2, hh, src, dst, m_agg);
    gru<<<N / 16, 128, 0, stream>>>(m_agg, h, hh, WihP, WhhP, b_ih, b_hh);
  }

  zero_f32<<<(G * HDIM + 255) / 256, 256, 0, stream>>>(gbuf, G * HDIM);
  pool<<<(N * HDIM + 255) / 256, 256, 0, stream>>>(h, batch, gbuf, N);
  readout<<<1, 256, 0, stream>>>(gbuf, W_r1, b_r1, W_r2, b_r2, out, G);
}
